// TopKRouter_4440996184650
// MI455X (gfx1250) — compile-verified, matrix operand order checked by reading, not run on hardware
//
#include <hip/hip_runtime.h>

// ---------------------------------------------------------------------------
// MoE Top-2 router for MI455X (gfx1250, wave32, WMMA).
// logits = x(16384x4096) @ gate_w(64x4096)^T via v_wmma_f32_16x16x32_bf16,
// then softmax / top-2 / aux-loss fused in LDS.
// Double-buffered LDS staging of gate_w, 64 K-columns per barrier,
// last chunk peeled so the steady-state loop is branch-free.
// ---------------------------------------------------------------------------

typedef __attribute__((ext_vector_type(16))) __bf16 v16bf;
typedef __attribute__((ext_vector_type(8)))  __bf16 v8bf;
typedef __attribute__((ext_vector_type(8)))  float  v8f;
typedef __attribute__((ext_vector_type(4)))  float  v4f;

#define D_DIM          4096
#define NEXP           64
#define TOKS_PER_BLOCK 128
#define THREADS        256
#define KCHUNK         64   // K columns staged per barrier (2 WMMA substeps)
#define SB_STRIDE      72   // bf16 per expert row per buffer (144B, 16B-aligned, bank-spread)
#define SL_STRIDE      65   // floats per token row in LDS (padded)

// Load one 32x16 bf16 B-fragment (column n = r of an expert tile) from LDS.
// v16bf element i maps to K = h*8 + i (i<8) and K = h*8 + 16 + (i-8) (i>=8),
// matching the CDNA5 16-bit A/B VGPR layout. Both chunks are 16B-aligned.
__device__ __forceinline__ v16bf load_bfrag(const __bf16* base) {
    v8bf lo = *(const v8bf*)(base);
    v8bf hi = *(const v8bf*)(base + 16);
    v16bf b;
#pragma unroll
    for (int i = 0; i < 8; ++i) { b[i] = lo[i]; b[i + 8] = hi[i]; }
    return b;
}

__global__ void router_init_ws(float* ws) {
    ws[threadIdx.x] = 0.0f;   // g_probsum[64] ++ g_count[64]
}

__global__ __launch_bounds__(THREADS) void moe_router_main(
    const float* __restrict__ x,       // (16384, 4096)
    const float* __restrict__ gw,      // (64, 4096)
    float* __restrict__ out_vals,      // (16384, 2)
    int*   __restrict__ out_idx,       // (16384, 2)
    float* __restrict__ g_probsum,     // (64,)
    float* __restrict__ g_count)       // (64,)
{
    __shared__ __bf16 sB[2][NEXP * SB_STRIDE];           // 18 KB double-buffered gate_w
    __shared__ float  sLog[TOKS_PER_BLOCK * SL_STRIDE];  // 32.5 KB logits tile
    __shared__ float  sInvDen[TOKS_PER_BLOCK];
    __shared__ int    sCnt[NEXP];

    const int tid  = threadIdx.x;
    const int wv   = tid >> 5;   // wave id 0..7 -> 16 tokens each
    const int lane = tid & 31;
    const int h    = lane >> 4;  // half-wave: selects K subset
    const int r    = lane & 15;  // A row / B column / D column

    if (tid < NEXP) sCnt[tid] = 0;

    v8f acc0 = {}, acc1 = {}, acc2 = {}, acc3 = {};

    const long tok0 = (long)blockIdx.x * TOKS_PER_BLOCK;
    const float* xrow = x + (tok0 + (long)(wv * 16 + r)) * D_DIM;

    // B staging assignment: 4 threads per expert row, 16 fp32 each.
    const int be = tid >> 2;           // expert 0..63
    const int bk = (tid & 3) * 16;     // k offset 0,16,32,48
    const float* gwp = gw + (long)be * D_DIM + bk;
    __bf16* sBw[2] = { &sB[0][be * SB_STRIDE + bk], &sB[1][be * SB_STRIDE + bk] };

    // Stage a 64-expert x 64-K gate_w chunk (fp32 -> bf16) into buffer `buf`.
    auto stage = [&](int buf, int kb2) {
        v4f f0 = *(const v4f*)(gwp + kb2);
        v4f f1 = *(const v4f*)(gwp + kb2 + 4);
        v4f f2 = *(const v4f*)(gwp + kb2 + 8);
        v4f f3 = *(const v4f*)(gwp + kb2 + 12);
        v8bf b0, b1;
#pragma unroll
        for (int i = 0; i < 4; ++i) {
            b0[i]     = (__bf16)f0[i];
            b0[i + 4] = (__bf16)f1[i];
            b1[i]     = (__bf16)f2[i];
            b1[i + 4] = (__bf16)f3[i];
        }
        *(v8bf*)(sBw[buf])     = b0;
        *(v8bf*)(sBw[buf] + 8) = b1;
    };

    // Compute both 32-K substeps of a staged 64-K chunk.
    auto compute = [&](int buf, int kb) {
#pragma unroll
        for (int s = 0; s < 2; ++s) {
            // A fragment: 16 fp32 of this lane's token row -> bf16
            const float* ap = xrow + kb + s * 32 + h * 8;
            v4f a0 = *(const v4f*)(ap);
            v4f a1 = *(const v4f*)(ap + 4);
            v4f a2 = *(const v4f*)(ap + 16);
            v4f a3 = *(const v4f*)(ap + 20);
            v16bf a;
#pragma unroll
            for (int i = 0; i < 4; ++i) {
                a[i]      = (__bf16)a0[i];
                a[i + 4]  = (__bf16)a1[i];
                a[i + 8]  = (__bf16)a2[i];
                a[i + 12] = (__bf16)a3[i];
            }

            // 4 expert tiles of 16: D(16x16) += A(16x32) x B(32x16)
            const __bf16* bb = &sB[buf][r * SB_STRIDE + s * 32 + h * 8];
            v16bf b0 = load_bfrag(bb + 0 * 16 * SB_STRIDE);
            v16bf b1 = load_bfrag(bb + 1 * 16 * SB_STRIDE);
            v16bf b2 = load_bfrag(bb + 2 * 16 * SB_STRIDE);
            v16bf b3 = load_bfrag(bb + 3 * 16 * SB_STRIDE);
            acc0 = __builtin_amdgcn_wmma_f32_16x16x32_bf16(false, a, false, b0, (short)0, acc0, false, false);
            acc1 = __builtin_amdgcn_wmma_f32_16x16x32_bf16(false, a, false, b1, (short)0, acc1, false, false);
            acc2 = __builtin_amdgcn_wmma_f32_16x16x32_bf16(false, a, false, b2, (short)0, acc2, false, false);
            acc3 = __builtin_amdgcn_wmma_f32_16x16x32_bf16(false, a, false, b3, (short)0, acc3, false, false);
        }
    };

    stage(0, 0);

    // Steady state: branch-free body; last chunk peeled.
    int kb = 0;
    for (; kb < D_DIM - KCHUNK; kb += KCHUNK) {
        const int buf = (kb >> 6) & 1;
        __syncthreads();                 // stores to `buf` visible; reads of `buf^1` retired
        stage(buf ^ 1, kb + KCHUNK);     // overlap next stage with this chunk's compute
        __builtin_prefetch(xrow + kb + KCHUNK, 0, 3);   // always in-bounds here
        compute(buf, kb);
    }
    {   // final chunk: nothing left to stage
        const int buf = (kb >> 6) & 1;
        __syncthreads();
        compute(buf, kb);
    }

    // Scatter D tiles to LDS: lane (h,r) VGPR i holds logits[M = i+8h][N = etile*16+r]
#pragma unroll
    for (int i = 0; i < 8; ++i) {
        float* row = &sLog[(wv * 16 + h * 8 + i) * SL_STRIDE];
        row[ 0 + r] = acc0[i];
        row[16 + r] = acc1[i];
        row[32 + r] = acc2[i];
        row[48 + r] = acc3[i];
    }
    __syncthreads();

    // Per-token softmax + top-2 (one thread per token)
    if (tid < TOKS_PER_BLOCK) {
        float* row = &sLog[tid * SL_STRIDE];
        float m = row[0];
#pragma unroll
        for (int e = 1; e < NEXP; ++e) m = fmaxf(m, row[e]);
        float sum = 0.0f, v0 = -1.0f, v1 = -1.0f;
        int   i0 = 0, i1 = 0;
#pragma unroll
        for (int e = 0; e < NEXP; ++e) {
            float p = __expf(row[e] - m);   // unnormalized prob; order == logits order
            row[e] = p;
            sum += p;
            if (p > v0)      { v1 = v0; i1 = i0; v0 = p; i0 = e; }
            else if (p > v1) { v1 = p;  i1 = e; }
        }
        float inv = 1.0f / sum;
        sInvDen[tid] = inv;
        long gt = tok0 + tid;
        out_vals[gt * 2 + 0] = v0 * inv;
        out_vals[gt * 2 + 1] = v1 * inv;
        out_idx [gt * 2 + 0] = i0;
        out_idx [gt * 2 + 1] = i1;
        atomicAdd(&sCnt[i0], 1);
        atomicAdd(&sCnt[i1], 1);
    }
    __syncthreads();

    // Per-expert partials for the aux loss (one thread per expert)
    if (tid < NEXP) {
        float s = 0.0f;
        for (int t = 0; t < TOKS_PER_BLOCK; ++t)
            s += sLog[t * SL_STRIDE + tid] * sInvDen[t];
        atomicAdd(&g_probsum[tid], s);
        atomicAdd(&g_count[tid], (float)sCnt[tid]);
    }
}

__global__ void router_loss_kernel(const float* __restrict__ g_probsum,
                                   const float* __restrict__ g_count,
                                   float* __restrict__ out_loss,
                                   float n_tokens)
{
    __shared__ float red[NEXP];
    const int e = threadIdx.x;
    float f = g_count[e] * (1.0f / (n_tokens * 2.0f));   // f_i
    float p = g_probsum[e] * (1.0f / n_tokens);          // p_i
    red[e] = f * p;                                      // E * mean == sum
    __syncthreads();
    for (int s = NEXP / 2; s > 0; s >>= 1) {
        if (e < s) red[e] += red[e + s];
        __syncthreads();
    }
    if (e == 0) *out_loss = red[0];
}

extern "C" void kernel_launch(void* const* d_in, const int* in_sizes, int n_in,
                              void* d_out, int out_size, void* d_ws, size_t ws_size,
                              hipStream_t stream) {
    const float* x  = (const float*)d_in[0];
    const float* gw = (const float*)d_in[1];
    const int n_tokens = in_sizes[0] / D_DIM;            // 16384

    float* out_vals = (float*)d_out;                     // n_tokens*2 floats
    int*   out_idx  = (int*)((float*)d_out + (size_t)n_tokens * 2);
    float* out_loss = (float*)d_out + (size_t)n_tokens * 4;

    float* g_probsum = (float*)d_ws;                     // 64 floats
    float* g_count   = g_probsum + NEXP;                 // 64 floats

    router_init_ws<<<1, 2 * NEXP, 0, stream>>>(g_probsum);
    moe_router_main<<<n_tokens / TOKS_PER_BLOCK, THREADS, 0, stream>>>(
        x, gw, out_vals, out_idx, g_probsum, g_count);
    router_loss_kernel<<<1, NEXP, 0, stream>>>(g_probsum, g_count, out_loss,
                                               (float)n_tokens);
}
